// StyleGANBlock_71528385348155
// MI455X (gfx1250) — matrix operand order repaired and checked by reading, not compiled, not run on hardware
//
#include <hip/hip_runtime.h>

#define C_IN 512
#define C_OUT 512
#define BATCH 8
#define STYLE_DIM 512
#define NPIX 4096            // 64*64 output pixels
#define KTOT (C_IN * 9)      // 4608

#define MT 128
#define NT 128
#define KT 32
#define KSTEPS (KTOT / KT)   // 144

typedef __bf16 bf16x16 __attribute__((ext_vector_type(16)));
typedef float  f32x8   __attribute__((ext_vector_type(8)));
typedef int    v4i     __attribute__((ext_vector_type(4)));
typedef __attribute__((address_space(1))) v4i* as1_v4i_p;
typedef __attribute__((address_space(3))) v4i* as3_v4i_p;

// ---- gfx1250 async global->LDS path (guarded; falls back to plain copy) ----
#if defined(__has_builtin)
#  if __has_builtin(__builtin_amdgcn_global_load_async_to_lds_b128)
#    define HAVE_ASYNC_B128 1
#  endif
#  if __has_builtin(__builtin_amdgcn_s_wait_asynccnt)
#    define HAVE_WAIT_ASYNC 1
#  endif
#endif

__device__ __forceinline__ void copy16_g2l(const void* g, void* l) {
#ifdef HAVE_ASYNC_B128
    __builtin_amdgcn_global_load_async_to_lds_b128(
        (as1_v4i_p)g, (as3_v4i_p)l, 0, 0);
#else
    *(uint4*)l = *(const uint4*)g;
#endif
}

__device__ __forceinline__ void wait_async_lds() {
#ifdef HAVE_ASYNC_B128
#  ifdef HAVE_WAIT_ASYNC
    __builtin_amdgcn_s_wait_asynccnt(0);
#  else
    asm volatile("s_wait_asynccnt 0x0" ::: "memory");
#  endif
#endif
}

__device__ __forceinline__ unsigned short f2bf(float f) {
    unsigned u = __float_as_uint(f);
    u += 0x7FFFu + ((u >> 16) & 1u);   // round-to-nearest-even
    return (unsigned short)(u >> 16);
}

// ---------------- Kernel 1: s[b][c] = style[b] . fc_w[c] + fc_b[c] ----------------
__global__ void style_fc_kernel(const float* __restrict__ style,
                                const float* __restrict__ fc_w,
                                const float* __restrict__ fc_b,
                                float* __restrict__ s) {
    int g = blockIdx.x * blockDim.x + threadIdx.x;   // 0..4095
    int b = g >> 9, c = g & 511;
    const float* st = style + b * STYLE_DIM;
    const float* w  = fc_w  + c * STYLE_DIM;
    float acc = fc_b[c];
    #pragma unroll 8
    for (int d = 0; d < STYLE_DIM; ++d) acc += st[d] * w[d];
    s[g] = acc;
}

// ---- Kernel 2: per (b,o): wmod = w*(s+1); d = rsqrt(sum wmod^2 + eps); emit bf16 ----
__global__ __launch_bounds__(256)
void modulate_kernel(const float* __restrict__ conv_w,
                     const float* __restrict__ s,
                     unsigned short* __restrict__ wbf) {
    __shared__ float red[256];
    int blk = blockIdx.x;            // b*512 + o
    int b = blk >> 9, o = blk & 511;
    int tid = threadIdx.x;
    const float* wrow = conv_w + (size_t)o * KTOT;
    const float* sb   = s + b * C_IN;

    float vals[18];
    float sum = 0.f;
    #pragma unroll
    for (int i = 0; i < 18; ++i) {
        int k = tid + i * 256;
        int c = k / 9;
        float wv = wrow[k] * (sb[c] + 1.0f);
        vals[i] = wv;
        sum += wv * wv;
    }
    red[tid] = sum;
    __syncthreads();
    for (int off = 128; off > 0; off >>= 1) {
        if (tid < off) red[tid] += red[tid + off];
        __syncthreads();
    }
    float d = rsqrtf(red[0] + 1e-8f);
    unsigned short* dst = wbf + (size_t)blk * KTOT;
    #pragma unroll
    for (int i = 0; i < 18; ++i) {
        int k = tid + i * 256;
        dst[k] = f2bf(vals[i] * d);
    }
}

// ---------------- Kernel 3: implicit-GEMM upsample+conv3x3 via WMMA ----------------
// Per batch b: Y[o][p] = sum_k A[o][k] * B[k][p],  k = c*9 + kh*3 + kw,
// B[k][p] gathered from x with the 2x nearest upsample folded into indexing.
// 128x128 output tile / WG, double-buffered LDS, async A staging, 8 WMMA / k-step / wave.
__global__ __launch_bounds__(256, 2)
void stylegan_conv_wmma_kernel(const float* __restrict__ x,
                               const unsigned short* __restrict__ wbf,
                               const float* __restrict__ noise,
                               const float* __restrict__ bias,
                               float* __restrict__ out) {
    __shared__ unsigned short As[2][MT][KT];    // 2 x 128x32 bf16 = 16KB  (A: [m][k])
    __shared__ unsigned short Bs[2][NT][KT];    // 2 x 128x32 bf16 = 16KB  (B^T: [n][k])

    const int b  = blockIdx.z;
    const int m0 = blockIdx.y * MT;
    const int n0 = blockIdx.x * NT;
    const int tid  = threadIdx.x;
    const int lane = tid & 31;
    const int wave = tid >> 5;
    const int wm = wave & 3;        // M base = wm*32 (two 16-row subtiles)
    const int wn = wave >> 2;       // N base = wn*64 (four 16-col subtiles)

    const int half = lane >> 4;
    const int l15  = lane & 15;

    f32x8 acc[2][4] = {};

    const unsigned short* wsrc = wbf + (size_t)(b * C_OUT + m0) * KTOT;
    const float* xb = x + (size_t)b * C_IN * 32 * 32;

    // A staging: 128 rows x 4 x 16B segments  (two b128 per thread per step)
    const int arow = tid >> 1;
    const int aoff = (tid & 1) * 16;
    // B staging: 128 n-rows x 2 k-halves of 16 (gather + cvt, two b128 ds stores)
    const int brow = tid >> 1;
    const int bkh  = (tid & 1) * 16;
    const int py = (n0 + brow) >> 6;
    const int px = (n0 + brow) & 63;

    auto stageA = [&](int k0, int buf) {
        copy16_g2l(wsrc + (size_t)arow * KTOT + k0 + aoff, &As[buf][arow][aoff]);
        copy16_g2l(wsrc + (size_t)arow * KTOT + k0 + aoff + 8, &As[buf][arow][aoff + 8]);
    };
    auto stageB = [&](int k0, int buf) {
        __attribute__((aligned(16))) unsigned short tmp[16];
        #pragma unroll
        for (int i = 0; i < 16; ++i) {
            int kg = k0 + bkh + i;
            int c  = kg / 9;
            int r  = kg - c * 9;
            int kh = r / 3;
            int kw = r - kh * 3;
            int yy = py + kh - 1;
            int xx = px + kw - 1;
            float v = 0.f;
            if ((unsigned)yy < 64u && (unsigned)xx < 64u)
                v = xb[(c * 32 + (yy >> 1)) * 32 + (xx >> 1)];
            tmp[i] = f2bf(v);
        }
        uint4* bd = (uint4*)&Bs[buf][brow][bkh];
        bd[0] = *(const uint4*)&tmp[0];
        bd[1] = *(const uint4*)&tmp[8];
    };

    stageA(0, 0);
    stageB(0, 0);

    int buf = 0;
    for (int ks = 0; ks < KSTEPS; ++ks) {
        wait_async_lds();          // A-tile async copies (this wave) complete
        __syncthreads();           // + everyone's ds stores visible

        if (ks + 1 < KSTEPS) {     // overlap next-slice staging with WMMA below
            stageA((ks + 1) * KT, buf ^ 1);
            stageB((ks + 1) * KT, buf ^ 1);
        }

        // per-lane fragments (contiguous k = 16*half + s, same order in A and B)
        union Frag { uint4 u[2]; bf16x16 v; };
        Frag af[2];
        #pragma unroll
        for (int mi = 0; mi < 2; ++mi) {
            af[mi].u[0] = *(const uint4*)&As[buf][wm * 32 + mi * 16 + l15][half * 16];
            af[mi].u[1] = *(const uint4*)&As[buf][wm * 32 + mi * 16 + l15][half * 16 + 8];
        }
        #pragma unroll
        for (int t = 0; t < 4; ++t) {
            Frag bfr;
            const uint4* bp = (const uint4*)&Bs[buf][wn * 64 + t * 16 + l15][half * 16];
            bfr.u[0] = bp[0];
            bfr.u[1] = bp[1];
            #pragma unroll
            for (int mi = 0; mi < 2; ++mi)
                acc[mi][t] = __builtin_amdgcn_wmma_f32_16x16x32_bf16(
                    false, af[mi].v, false, bfr.v, (short)0, acc[mi][t], false, false);
        }
        buf ^= 1;
    }

    // ---- epilogue: C-layout (lane = N, VGPR v -> M = half*8 + v); bias+noise+leaky
    #pragma unroll
    for (int t = 0; t < 4; ++t) {
        int n = n0 + wn * 64 + t * 16 + l15;
        float nz = noise[b * NPIX + n];
        #pragma unroll
        for (int mi = 0; mi < 2; ++mi) {
            #pragma unroll
            for (int v = 0; v < 8; ++v) {
                int o = m0 + wm * 32 + mi * 16 + half * 8 + v;
                float y = acc[mi][t][v] + bias[o] + nz;
                out[(size_t)(b * C_OUT + o) * NPIX + n] = (y > 0.f) ? y : 0.2f * y;
            }
        }
    }
}

extern "C" void kernel_launch(void* const* d_in, const int* in_sizes, int n_in,
                              void* d_out, int out_size, void* d_ws, size_t ws_size,
                              hipStream_t stream) {
    const float* x      = (const float*)d_in[0];
    const float* style  = (const float*)d_in[1];
    const float* noise  = (const float*)d_in[2];
    const float* conv_w = (const float*)d_in[3];
    const float* fc_w   = (const float*)d_in[4];
    const float* fc_b   = (const float*)d_in[5];
    const float* bias   = (const float*)d_in[6];
    float* out = (float*)d_out;

    // workspace layout: s (8*512 f32 = 16KB) | modulated bf16 weights (8*512*4608*2B)
    float* s_mod = (float*)d_ws;
    unsigned short* wbf = (unsigned short*)((char*)d_ws + 16 * 1024);

    style_fc_kernel<<<16, 256, 0, stream>>>(style, fc_w, fc_b, s_mod);
    modulate_kernel<<<BATCH * C_OUT, 256, 0, stream>>>(conv_w, s_mod, wbf);

    dim3 grid(NPIX / NT, C_OUT / MT, BATCH);   // (32, 4, 8)
    stylegan_conv_wmma_kernel<<<grid, 256, 0, stream>>>(x, wbf, noise, bias, out);
}
